// CausalPrefixAttention_5987184411323
// MI455X (gfx1250) — matrix operand order repaired and checked by reading, not compiled, not run on hardware
//
#include <hip/hip_runtime.h>
#include <cstdint>
#include <cstddef>

// ---------------------------------------------------------------------------
// CDNA5 (gfx1250) prefix-causal attention block.
// Matmuls via V_WMMA_F32_16X16X32_F16; global->LDS staging of non-transposed
// tiles via GLOBAL_LOAD_ASYNC_TO_LDS_B128 (ASYNCcnt) with double buffering.
// Shapes fixed by the reference: B=2, L=2048, D=1024, H=16, d=64, N_qkv=3072.
// ---------------------------------------------------------------------------

typedef __attribute__((ext_vector_type(16))) _Float16     v16h;
typedef __attribute__((ext_vector_type(8)))  float        v8f;
typedef __attribute__((ext_vector_type(4)))  unsigned int v4u;

union AFrag { v16h v; v4u q[2]; uint32_t u[8]; _Float16 h[16]; };
union BFrag { v16h v; v4u q[2]; uint32_t u[8]; _Float16 h[16]; };
union CFrag { v8f  v; float f[8]; };

__device__ __forceinline__ void async_lds_b128(const _Float16* gsrc, void* lds) {
  uint32_t off = (uint32_t)(uintptr_t)lds;   // generic addr low 32 bits == LDS offset
  asm volatile("global_load_async_to_lds_b128 %0, %1, off"
               :: "v"(off), "v"(gsrc) : "memory");
}
__device__ __forceinline__ void async_lds_b128x2(const _Float16* gsrc, void* lds) {
  uint32_t off = (uint32_t)(uintptr_t)lds;
  asm volatile("global_load_async_to_lds_b128 %0, %1, off\n\t"
               "global_load_async_to_lds_b128 %0, %1, off offset:16"
               :: "v"(off), "v"(gsrc) : "memory");
}
__device__ __forceinline__ void wait_asynccnt0() {
  asm volatile("s_wait_asynccnt 0" ::: "memory");
}

// ---------------------------------------------------------------------------
// f32 -> f16 conversion
// ---------------------------------------------------------------------------
__global__ void cvt_f32_to_f16(const float* __restrict__ in,
                               _Float16* __restrict__ out, int n) {
  int i = blockIdx.x * blockDim.x + threadIdx.x;
  int stride = gridDim.x * blockDim.x;
  for (; i < n; i += stride) out[i] = (_Float16)in[i];
}

// ---------------------------------------------------------------------------
// Tiled WMMA GEMM: C[M,N] = A[M,K] * B[K,N], f16 in, f32 accumulate.
// 256 threads (8 waves), tile 64x64, BK=32, double-buffered LDS:
//   A tile  : async global->LDS (b128 per thread)
//   B tile  : global b128 -> regs (prefetched one step ahead) -> transposed LDS
// mode 0: scatter f16 results into q/k/v [B=2,H=16,L=2048,64]
// mode 1: plain f32 store to out32[M,N]
// ---------------------------------------------------------------------------
__global__ __launch_bounds__(256) void gemm_wmma_f16(
    const _Float16* __restrict__ A, const _Float16* __restrict__ Bm,
    int M, int N, int K, int mode,
    _Float16* __restrict__ qd, _Float16* __restrict__ kd,
    _Float16* __restrict__ vd, float* __restrict__ out32) {
  __shared__ alignas(16) _Float16 As[2][64 * 48];  // [row][k], stride 48
  __shared__ alignas(16) _Float16 Bs[2][64 * 40];  // transposed [n][k], stride 40

  const int tid = threadIdx.x;
  const int w   = tid >> 5;
  const int l   = tid & 31;
  const int lo  = l & 15;
  const int hi  = l >> 4;
  const int wave_m = w >> 1;   // 0..3 -> 16-row slice
  const int wave_n = w & 1;    // 0..1 -> 32-col slice

  const int m0 = blockIdx.y * 64;
  const int n0 = blockIdx.x * 64;

  CFrag acc[2];
#pragma unroll
  for (int i = 0; i < 8; ++i) { acc[0].f[i] = 0.f; acc[1].f[i] = 0.f; }

  const int arow = tid >> 2;            // 0..63
  const int acol = (tid & 3) * 8;       // 0,8,16,24
  const int brow = tid >> 3;            // 0..31
  const int bcol = (tid & 7) * 8;       // 0..56

  union { v4u u4; _Float16 h[8]; } breg;

  // prologue: stage tile 0
  async_lds_b128(A + (size_t)(m0 + arow) * K + acol, &As[0][arow * 48 + acol]);
  breg.u4 = *(const v4u*)(Bm + (size_t)brow * N + n0 + bcol);

  int cur = 0;
  for (int k0 = 0; k0 < K; k0 += 32) {
    // finish staging of current tile
    wait_asynccnt0();
#pragma unroll
    for (int i = 0; i < 8; ++i) Bs[cur][(bcol + i) * 40 + brow] = breg.h[i];
    __syncthreads();

    // prefetch next tile while computing on this one
    const int k1 = k0 + 32;
    if (k1 < K) {
      async_lds_b128(A + (size_t)(m0 + arow) * K + k1 + acol,
                     &As[cur ^ 1][arow * 48 + acol]);
      breg.u4 = *(const v4u*)(Bm + (size_t)(k1 + brow) * N + n0 + bcol);
    }

    // A fragment (rows m0 + wave_m*16 .. +15): two contiguous b128 LDS reads
    AFrag a;
    const uint32_t* as32 = (const uint32_t*)As[cur];
    const int abase = (wave_m * 16 + lo) * 24;  // stride 48 f16 = 24 words
    a.q[0] = *(const v4u*)(as32 + abase + hi * 4);
    a.q[1] = *(const v4u*)(as32 + abase + 8 + hi * 4);

    const uint32_t* bs32 = (const uint32_t*)Bs[cur];
#pragma unroll
    for (int nn = 0; nn < 2; ++nn) {
      BFrag b;
      const int bbase = (wave_n * 32 + nn * 16 + lo) * 20;  // stride 40 f16
      b.q[0] = *(const v4u*)(bs32 + bbase + hi * 4);
      b.q[1] = *(const v4u*)(bs32 + bbase + 8 + hi * 4);
      acc[nn].v = __builtin_amdgcn_wmma_f32_16x16x32_f16(
          false, a.v, false, b.v, (short)0, acc[nn].v, false, false);
    }
    cur ^= 1;
  }

  // epilogue
#pragma unroll
  for (int nn = 0; nn < 2; ++nn) {
#pragma unroll
    for (int r = 0; r < 8; ++r) {
      const int rt = m0 + wave_m * 16 + r + hi * 8;
      const int ct = n0 + wave_n * 32 + nn * 16 + lo;
      const float val = acc[nn].f[r];
      if (mode == 0) {
        const int s  = ct >> 10;
        const int rm = ct & 1023;
        const int h  = rm >> 6;
        const int dd = rm & 63;
        const int b  = rt >> 11;
        const int tk = rt & 2047;
        _Float16* dst = (s == 0) ? qd : ((s == 1) ? kd : vd);
        dst[(((size_t)(b * 16 + h) * 2048 + tk) << 6) + dd] = (_Float16)val;
      } else {
        out32[(size_t)rt * N + ct] = val;
      }
    }
  }
}

// ---------------------------------------------------------------------------
// Flash attention with prefix-causal mask.
// Grid: (L/64, B*H). Block: 128 threads = 4 waves; wave w owns 16 query rows.
// K tile staged via async global->LDS; V tile transposed via VGPR scatter.
// ---------------------------------------------------------------------------
__global__ __launch_bounds__(128) void flash_attn_wmma(
    const _Float16* __restrict__ q, const _Float16* __restrict__ kk_,
    const _Float16* __restrict__ vv_, _Float16* __restrict__ o,
    const int* __restrict__ nprefix) {
  __shared__ alignas(16) _Float16 Kt[32 * 72];     // [key][d], stride 72
  __shared__ alignas(16) _Float16 Vt[64 * 40];     // transposed [d][key], stride 40
  __shared__ alignas(16) _Float16 Ps[4][16 * 40];  // per-wave P tile, stride 40

  const int tid = threadIdx.x;
  const int w   = tid >> 5;
  const int l   = tid & 31;
  const int lo  = l & 15;
  const int hi  = l >> 4;

  const int bh = blockIdx.y;             // 0..31
  const int q0 = blockIdx.x * 64;
  const int Mp = *nprefix;
  const float scale = 0.125f;            // 1/sqrt(64)

  const size_t base = (size_t)bh * 2048 * 64;

  // preload Q A-fragments (two 32-wide K slabs over d=64) from global
  AFrag qa[2];
  {
    const int tok = q0 + w * 16 + lo;
    const uint32_t* q32 = (const uint32_t*)(q + base + (size_t)tok * 64);
#pragma unroll
    for (int ks = 0; ks < 2; ++ks) {
      qa[ks].q[0] = *(const v4u*)(q32 + ks * 16 + hi * 4);
      qa[ks].q[1] = *(const v4u*)(q32 + ks * 16 + 8 + hi * 4);
    }
  }

  float mrow[8], lsum[8];
  CFrag Ofr[4];
#pragma unroll
  for (int r = 0; r < 8; ++r) { mrow[r] = -1e30f; lsum[r] = 0.f; }
#pragma unroll
  for (int nn = 0; nn < 4; ++nn)
#pragma unroll
    for (int r = 0; r < 8; ++r) Ofr[nn].f[r] = 0.f;

  const int kend = q0 + 64;         // covers causal band + prefix
  const int keyr = tid >> 2;        // 0..31
  const int c16  = (tid & 3) * 16;  // 0,16,32,48

  for (int kb = 0; kb < kend; kb += 32) {
    // ---- stage K async (2x b128/thread), V transposed via regs ----
    async_lds_b128x2(kk_ + base + (size_t)(kb + keyr) * 64 + c16,
                     &Kt[keyr * 72 + c16]);
    {
      union { v4u u4[2]; _Float16 h[16]; } tv;
      const v4u* vs = (const v4u*)(vv_ + base + (size_t)(kb + keyr) * 64 + c16);
      tv.u4[0] = vs[0];
      tv.u4[1] = vs[1];
#pragma unroll
      for (int i = 0; i < 16; ++i) Vt[(c16 + i) * 40 + keyr] = tv.h[i];
    }
    wait_asynccnt0();
    __syncthreads();

    // ---- S = Q * K^T  (16 x 32 scores per wave) ----
    CFrag s0, s1;
#pragma unroll
    for (int r = 0; r < 8; ++r) { s0.f[r] = 0.f; s1.f[r] = 0.f; }
    const uint32_t* kt32 = (const uint32_t*)Kt;
#pragma unroll
    for (int ks = 0; ks < 2; ++ks) {
      BFrag b0, b1;
      const int kofs = ks * 16 + hi * 4;           // word offset within a key row
      b0.q[0] = *(const v4u*)(kt32 + lo * 36 + kofs);
      b0.q[1] = *(const v4u*)(kt32 + lo * 36 + kofs + 8);
      b1.q[0] = *(const v4u*)(kt32 + (16 + lo) * 36 + kofs);
      b1.q[1] = *(const v4u*)(kt32 + (16 + lo) * 36 + kofs + 8);
      s0.v = __builtin_amdgcn_wmma_f32_16x16x32_f16(
          false, qa[ks].v, false, b0.v, (short)0, s0.v, false, false);
      s1.v = __builtin_amdgcn_wmma_f32_16x16x32_f16(
          false, qa[ks].v, false, b1.v, (short)0, s1.v, false, false);
    }

    // ---- mask + scale, online softmax ----
    bool al0[8], al1[8];
#pragma unroll
    for (int r = 0; r < 8; ++r) {
      const int qi  = q0 + w * 16 + r + hi * 8;
      const int kj0 = kb + lo;
      const int kj1 = kb + 16 + lo;
      al0[r] = (kj0 <= qi) || (qi >= Mp && kj0 < Mp);
      al1[r] = (kj1 <= qi) || (qi >= Mp && kj1 < Mp);
      s0.f[r] = al0[r] ? s0.f[r] * scale : -1e30f;
      s1.f[r] = al1[r] ? s1.f[r] * scale : -1e30f;
    }
#pragma unroll
    for (int r = 0; r < 8; ++r) {
      float vmx = fmaxf(s0.f[r], s1.f[r]);
#pragma unroll
      for (int off = 1; off < 16; off <<= 1)
        vmx = fmaxf(vmx, __shfl_xor(vmx, off, 32));
      const float mnew  = fmaxf(mrow[r], vmx);
      const float alpha = __expf(mrow[r] - mnew);
      mrow[r] = mnew;
      const float p0 = al0[r] ? __expf(s0.f[r] - mnew) : 0.f;
      const float p1 = al1[r] ? __expf(s1.f[r] - mnew) : 0.f;
      s0.f[r] = p0;
      s1.f[r] = p1;
      float ps = p0 + p1;
#pragma unroll
      for (int off = 1; off < 16; off <<= 1)
        ps += __shfl_xor(ps, off, 32);
      lsum[r] = lsum[r] * alpha + ps;
#pragma unroll
      for (int nn = 0; nn < 4; ++nn) Ofr[nn].f[r] *= alpha;
    }

    // ---- relayout P: C-layout regs -> LDS -> A-fragment ----
    {
      _Float16* pw = Ps[w];
#pragma unroll
      for (int r = 0; r < 8; ++r) {
        pw[(r + hi * 8) * 40 + lo]      = (_Float16)s0.f[r];
        pw[(r + hi * 8) * 40 + 16 + lo] = (_Float16)s1.f[r];
      }
    }
    AFrag pa;
    {
      const uint32_t* ps32 = (const uint32_t*)Ps[w];
      pa.q[0] = *(const v4u*)(ps32 + lo * 20 + hi * 4);
      pa.q[1] = *(const v4u*)(ps32 + lo * 20 + 8 + hi * 4);
    }

    // ---- O += P * V ----
    const uint32_t* vt32 = (const uint32_t*)Vt;
#pragma unroll
    for (int nn = 0; nn < 4; ++nn) {
      BFrag bv;
      const int vbase = (nn * 16 + lo) * 20;
      bv.q[0] = *(const v4u*)(vt32 + vbase + hi * 4);
      bv.q[1] = *(const v4u*)(vt32 + vbase + 8 + hi * 4);
      Ofr[nn].v = __builtin_amdgcn_wmma_f32_16x16x32_f16(
          false, pa.v, false, bv.v, (short)0, Ofr[nn].v, false, false);
    }
    __syncthreads();
  }

  // ---- normalize + store (head-concat layout [B*L, 1024]) ----
  const int b = bh >> 4;
  const int h = bh & 15;
#pragma unroll
  for (int r = 0; r < 8; ++r) {
    const float inv = 1.0f / lsum[r];
    const int qi = q0 + w * 16 + r + hi * 8;
    const size_t rowoff = ((size_t)(b * 2048 + qi)) * 1024 + h * 64;
#pragma unroll
    for (int nn = 0; nn < 4; ++nn)
      o[rowoff + nn * 16 + lo] = (_Float16)(Ofr[nn].f[r] * inv);
  }
}

// ---------------------------------------------------------------------------
// Host-side orchestration
// ---------------------------------------------------------------------------
extern "C" void kernel_launch(void* const* d_in, const int* in_sizes, int n_in,
                              void* d_out, int out_size, void* d_ws, size_t ws_size,
                              hipStream_t stream) {
  (void)in_sizes; (void)n_in; (void)out_size; (void)ws_size;

  const float* x      = (const float*)d_in[0];   // [2,2048,1024]
  const float* w_qkv  = (const float*)d_in[1];   // [1024,3072]
  const float* w_proj = (const float*)d_in[2];   // [1024,1024]
  const int*   npref  = (const int*)d_in[3];     // scalar

  const int M = 4096;        // B*L
  const int D = 1024;
  const int Nqkv = 3072;

  uint8_t* ws = (uint8_t*)d_ws;
  _Float16* x16     = (_Float16*)(ws);                       // 8 MiB
  _Float16* wqkv16  = (_Float16*)(ws + ((size_t)8  << 20));  // 6 MiB
  _Float16* wproj16 = (_Float16*)(ws + ((size_t)14 << 20));  // 2 MiB
  _Float16* q16     = (_Float16*)(ws + ((size_t)16 << 20));  // 8 MiB
  _Float16* k16     = (_Float16*)(ws + ((size_t)24 << 20));  // 8 MiB
  _Float16* v16     = (_Float16*)(ws + ((size_t)32 << 20));  // 8 MiB
  _Float16* att16   = (_Float16*)(ws + ((size_t)40 << 20));  // 8 MiB

  cvt_f32_to_f16<<<2048, 256, 0, stream>>>(x,      x16,     M * D);
  cvt_f32_to_f16<<<2048, 256, 0, stream>>>(w_qkv,  wqkv16,  D * Nqkv);
  cvt_f32_to_f16<<<1024, 256, 0, stream>>>(w_proj, wproj16, D * D);

  {
    dim3 grid(Nqkv / 64, M / 64);
    gemm_wmma_f16<<<grid, 256, 0, stream>>>(x16, wqkv16, M, Nqkv, D, /*mode=*/0,
                                            q16, k16, v16, nullptr);
  }
  {
    dim3 grid(2048 / 64, 32 /* B*H */);
    flash_attn_wmma<<<grid, 128, 0, stream>>>(q16, k16, v16, att16, npref);
  }
  {
    dim3 grid(D / 64, M / 64);
    gemm_wmma_f16<<<grid, 256, 0, stream>>>(att16, wproj16, M, D, D, /*mode=*/1,
                                            nullptr, nullptr, nullptr,
                                            (float*)d_out);
  }
}
